// VitDecoder_47193100648833
// MI455X (gfx1250) — compile-verified, matrix operand order checked by reading, not run on hardware
//
#include <hip/hip_runtime.h>
#include <hip/hip_bf16.h>

#define DEPTH 4
#define D 768
#define NHEAD 12
#define HD 64
#define BB 8
#define NQ 1024
#define NC 1024
#define FF (4 * D)
#define M_ROWS (BB * NQ)    // 8192
#define CTX_ROWS (BB * NC)  // 8192
#define SPAD 40             // LDS row pitch (elements): 80B, 16B-aligned, bank-safe
#define KPAD 72             // LDS pitch for 64-wide K rows: 144B, 16B-aligned

typedef __bf16 bf16;
typedef __attribute__((ext_vector_type(16))) __bf16 v16bf;
typedef __attribute__((ext_vector_type(8)))  __bf16 v8bf;
typedef __attribute__((ext_vector_type(8)))  float  v8f;

// ---------------------------------------------------------------------------
// WMMA helpers
// ---------------------------------------------------------------------------
static __device__ __forceinline__ v8f wmma_bf16(v16bf a, v16bf b, v8f c) {
  // D(16x16,f32) = A(16x32,bf16) * B(32x16,bf16) + C
  return __builtin_amdgcn_wmma_f32_16x16x32_bf16(
      /*neg_a=*/false, a, /*neg_b=*/false, b,
      /*c_mod=*/(short)0, c, /*reuse_a=*/false, /*reuse_b=*/false);
}

// Load a 16x32 bf16 fragment from a row-major tile (row stride `ld` elements).
// Works for BOTH the A-operand ([16 M-rows x 32 K]) and the B-operand
// ([32 K x 16 N] == rows of W[N,K]): lane L holds row (L&15), K-base (L>>4)*8,
// elements 0..7 = K+0..7, elements 8..15 = K+16..23  (per CDNA5 ISA 7.12.2).
static __device__ __forceinline__ v16bf load_frag(const bf16* __restrict__ base,
                                                  int ld) {
  const int lane = threadIdx.x & 31;
  const bf16* p = base + (size_t)(lane & 15) * ld + ((lane >> 4) << 3);
  union { v16bf v; v8bf h[2]; } u;
  u.h[0] = *(const v8bf*)(p);
  u.h[1] = *(const v8bf*)(p + 16);
  return u.v;
}

// Same fragment pattern from a per-lane resolved LDS pointer (row/K-base
// already folded in by the caller); halves at +0 and +16 elements.
// (InferAddressSpaces turns these into ds_load_b128.)
static __device__ __forceinline__ v16bf lds_frag(const bf16* p) {
  union { v16bf v; v8bf h[2]; } u;
  u.h[0] = *(const v8bf*)(p);
  u.h[1] = *(const v8bf*)(p + 16);
  return u.v;
}

static __device__ __forceinline__ float lane_bcast(float v, int src_lane) {
  return __int_as_float(
      __builtin_amdgcn_ds_bpermute(src_lane << 2, __float_as_int(v)));
}

// CDNA5 async Global->LDS copy (per-lane 16B), tracked by ASYNCcnt.
static __device__ __forceinline__ void async_load_b128(unsigned int lds_addr,
                                                       const bf16* gptr) {
  asm volatile("global_load_async_to_lds_b128 %0, %1, off"
               :
               : "v"(lds_addr), "v"((unsigned long long)(uintptr_t)gptr)
               : "memory");
}

// ---------------------------------------------------------------------------
// GEMM:  C[M,N] = A[M,K](bf16) * W[N,K]^T(bf16) + bias
// epi: 0 = store bf16, 1 = outf += (residual add, fp32), 2 = exact GELU->bf16
// Block: 256 threads (8 waves). Block tile 128x64, wave tile 32x32.
// Double-buffered async Global->LDS staging of both operand tiles.
// ---------------------------------------------------------------------------
__global__ void __launch_bounds__(256)
gemm_wmma(const bf16* __restrict__ A, int lda,
          const bf16* __restrict__ W,               // [N,K], ldw == K
          const float* __restrict__ bias,
          bf16* __restrict__ outb, float* __restrict__ outf, int ldo,
          int K, int epi) {
  __shared__ __align__(16) bf16 sA[2][128][SPAD];
  __shared__ __align__(16) bf16 sB[2][64][SPAD];

  const int tid = threadIdx.x;
  const int lane = tid & 31;
  const int wid = tid >> 5;
  const int wm = wid & 3;       // 0..3  (M direction)
  const int wn = wid >> 2;      // 0..1  (N direction)
  const int rowblk = blockIdx.y * 128;
  const int colblk = blockIdx.x * 64;

  // Producer mapping: thread t fills 16B segment (t&3) of rows t>>2 (A: two
  // rows, +0 and +64; B: one row). Advances 32 elements (64B) per K-step.
  const int prow = tid >> 2;           // 0..63
  const int pseg = (tid & 3) * 8;      // element offset within 32-wide row
  const bf16* gA0 = A + (size_t)(rowblk + prow) * lda + pseg;
  const bf16* gA1 = A + (size_t)(rowblk + 64 + prow) * lda + pseg;
  const bf16* gB  = W + (size_t)(colblk + prow) * K + pseg;
  unsigned int lA0[2], lA1[2], lB[2];
#pragma unroll
  for (int b = 0; b < 2; ++b) {
    lA0[b] = (unsigned int)(uintptr_t)&sA[b][prow][pseg];
    lA1[b] = (unsigned int)(uintptr_t)&sA[b][64 + prow][pseg];
    lB[b]  = (unsigned int)(uintptr_t)&sB[b][prow][pseg];
  }

  const int fr  = lane & 15;           // fragment row within 16-row tile
  const int fc  = (lane >> 4) << 3;    // fragment K-base (0 or 8)

  v8f acc[2][2] = {{{0.f}, {0.f}}, {{0.f}, {0.f}}};

  // stage 0
  async_load_b128(lA0[0], gA0);
  async_load_b128(lA1[0], gA1);
  async_load_b128(lB[0],  gB);

  const int ksteps = K >> 5;
  for (int ks = 0; ks < ksteps; ++ks) {
    const int cur = ks & 1;
    if (ks + 1 < ksteps) {
      const int knext = (ks + 1) << 5;
      async_load_b128(lA0[cur ^ 1], gA0 + knext);
      async_load_b128(lA1[cur ^ 1], gA1 + knext);
      async_load_b128(lB[cur ^ 1],  gB + knext);
      asm volatile("s_wait_asynccnt 0x3" ::: "memory");  // stage `cur` landed
    } else {
      asm volatile("s_wait_asynccnt 0x0" ::: "memory");
    }
    __syncthreads();

    v16bf a0 = lds_frag(&sA[cur][wm * 32 + fr][fc]);
    v16bf a1 = lds_frag(&sA[cur][wm * 32 + 16 + fr][fc]);
    v16bf b0 = lds_frag(&sB[cur][wn * 32 + fr][fc]);
    v16bf b1 = lds_frag(&sB[cur][wn * 32 + 16 + fr][fc]);
    acc[0][0] = wmma_bf16(a0, b0, acc[0][0]);
    acc[0][1] = wmma_bf16(a0, b1, acc[0][1]);
    acc[1][0] = wmma_bf16(a1, b0, acc[1][0]);
    acc[1][1] = wmma_bf16(a1, b1, acc[1][1]);

    __syncthreads();  // all reads of `cur` done before it is refilled
  }

  const int row0 = rowblk + wm * 32;
  const int col0 = colblk + wn * 32;
  const int nl = lane & 15;            // local N (column) index
  const int mb = (lane >> 4) << 3;     // local M base (0 or 8)
#pragma unroll
  for (int mi = 0; mi < 2; ++mi) {
#pragma unroll
    for (int ni = 0; ni < 2; ++ni) {
      const int col = col0 + ni * 16 + nl;
      const float bv = bias[col];
#pragma unroll
      for (int v = 0; v < 8; ++v) {
        const int row = row0 + mi * 16 + mb + v;
        const float val = acc[mi][ni][v] + bv;
        const size_t idx = (size_t)row * ldo + col;
        if (epi == 0) {
          outb[idx] = (bf16)val;
        } else if (epi == 1) {
          outf[idx] += val;
        } else {
          outb[idx] = (bf16)(0.5f * val * (1.0f + erff(val * 0.70710678118f)));
        }
      }
    }
  }
}

// ---------------------------------------------------------------------------
// Fused attention (flash-style, online softmax), one (b, head) per blockIdx.yz,
// 8 waves/block, 16 queries per wave. Computes S^T = K*Q^T so the softmax'd
// probabilities land directly in the A-fragment layout for P*V.
// The shared K tile (32x64) and V^T tile (64x32) per 32-key chunk are staged
// into LDS with double-buffered async copies (all 8 waves consume the same
// tiles; only Q differs per wave).
//   q rows:   qbase[(b*NQ+q)*sq + h*HD + :]
//   k rows:   kbase[(b*NC+k)*sk + h*HD + :]
//   vT:       vt[((b*NHEAD+h)*HD + d)*NC + k]
//   out:      out[(b*NQ+q)*so + h*HD + d]   (bf16)
// ---------------------------------------------------------------------------
__global__ void __launch_bounds__(256)
attn_wmma(const bf16* __restrict__ qbase, int sq,
          const bf16* __restrict__ kbase, int sk,
          const bf16* __restrict__ vt,
          bf16* __restrict__ out, int so, float scale) {
  __shared__ __align__(16) bf16 sK[2][32][KPAD];   // 32 keys x 64 hd
  __shared__ __align__(16) bf16 sV[2][64][SPAD];   // 64 d    x 32 k

  const int b = blockIdx.z;
  const int h = blockIdx.y;
  const int tid = threadIdx.x;
  const int wid = tid >> 5;
  const int lane = tid & 31;
  const int qt = (blockIdx.x * 8 + wid) * 16;  // query tile origin

  const bf16* qp = qbase + (size_t)(b * NQ + qt) * sq + h * HD;
  const bf16* kp = kbase + (size_t)(b * NC) * sk + h * HD;
  const bf16* vtp = vt + (size_t)((b * NHEAD + h) * HD) * NC;

  // Producer mapping (per 32-key stage, 16B per async transfer):
  //   K tile:  thread t -> key row t>>3 (0..31), segment t&7 of 128B row
  //   V tile:  thread t -> d   row t>>2 (0..63), segment t&3 of 64B row
  const int krow = tid >> 3, kseg = (tid & 7) * 8;
  const int vrow = tid >> 2, vseg = (tid & 3) * 8;
  const bf16* gK = kp + (size_t)krow * sk + kseg;   // advance by 32*sk / stage
  const bf16* gV = vtp + (size_t)vrow * NC + vseg;  // advance by 32 / stage
  unsigned int lK[2], lV[2];
#pragma unroll
  for (int bb = 0; bb < 2; ++bb) {
    lK[bb] = (unsigned int)(uintptr_t)&sK[bb][krow][kseg];
    lV[bb] = (unsigned int)(uintptr_t)&sV[bb][vrow][vseg];
  }

  // Q tile as two B-fragments (hd chunks 0..31, 32..63), resident all loop.
  const v16bf qf0 = load_frag(qp, sq);
  const v16bf qf1 = load_frag(qp + 32, sq);

  v8f o[4] = {{0.f}, {0.f}, {0.f}, {0.f}};  // 16q x 64d accumulator (4 tiles)
  float mrun = -1e30f;   // running max for query q = lane&15
  float lrun = 0.f;      // running denom for query q = lane&15
  const int mb = (lane >> 4) << 3;
  const int fr = lane & 15;
  const int fc = (lane >> 4) << 3;

  // stage 0
  async_load_b128(lK[0], gK);
  async_load_b128(lV[0], gV);

  const int nsteps = NC >> 5;
  for (int ks = 0; ks < nsteps; ++ks) {
    const int cur = ks & 1;
    if (ks + 1 < nsteps) {
      const int kc1 = (ks + 1) << 5;
      async_load_b128(lK[cur ^ 1], gK + (size_t)kc1 * sk);
      async_load_b128(lV[cur ^ 1], gV + kc1);
      asm volatile("s_wait_asynccnt 0x2" ::: "memory");  // stage `cur` landed
    } else {
      asm volatile("s_wait_asynccnt 0x0" ::: "memory");
    }
    __syncthreads();

    // S^T tiles: c0 = keys 0..15, c1 = keys 16..31 of this chunk (M = keys).
    v16bf ka0 = lds_frag(&sK[cur][fr][fc]);
    v16bf ka1 = lds_frag(&sK[cur][fr][32 + fc]);
    v16bf kb0 = lds_frag(&sK[cur][16 + fr][fc]);
    v16bf kb1 = lds_frag(&sK[cur][16 + fr][32 + fc]);
    v8f c0 = {0.f}, c1 = {0.f};
    c0 = wmma_bf16(ka0, qf0, c0);
    c0 = wmma_bf16(ka1, qf1, c0);
    c1 = wmma_bf16(kb0, qf0, c1);
    c1 = wmma_bf16(kb1, qf1, c1);

    // Chunk max over 32 keys for this lane's query (8+8 in-lane, then xor16).
    float cmax = -1e30f;
#pragma unroll
    for (int v = 0; v < 8; ++v) {
      c0[v] *= scale; c1[v] *= scale;
      cmax = fmaxf(cmax, fmaxf(c0[v], c1[v]));
    }
    cmax = fmaxf(cmax, lane_bcast(cmax, lane ^ 16));
    const float mnew = fmaxf(mrun, cmax);
    const float alpha = __expf(mrun - mnew);
    mrun = mnew;

    // exp; P lands exactly in the A-fragment layout for P*V.
    union { v16bf v; v8bf hh[2]; } pu;
    float lsum = 0.f;
#pragma unroll
    for (int v = 0; v < 8; ++v) {
      const float e0 = __expf(c0[v] - mnew);
      const float e1 = __expf(c1[v] - mnew);
      lsum += e0 + e1;
      pu.hh[0][v] = (bf16)e0;   // elements 0..7  <-> chunk keys mb + v
      pu.hh[1][v] = (bf16)e1;   // elements 8..15 <-> chunk keys 16 + mb + v
    }
    lsum += lane_bcast(lsum, lane ^ 16);
    lrun = lrun * alpha + lsum;

    // Rescale O accumulator: per-element alpha for q = mb + v (lives in lane q)
    float al[8];
#pragma unroll
    for (int v = 0; v < 8; ++v) al[v] = lane_bcast(alpha, mb + v);
#pragma unroll
    for (int t = 0; t < 4; ++t)
#pragma unroll
      for (int v = 0; v < 8; ++v) o[t][v] *= al[v];

    // O += P * V  (V^T rows contiguous in k => b128 LDS fragment loads)
#pragma unroll
    for (int t = 0; t < 4; ++t) {
      v16bf vf = lds_frag(&sV[cur][t * 16 + fr][fc]);
      o[t] = wmma_bf16(pu.v, vf, o[t]);
    }

    __syncthreads();  // all reads of `cur` done before it is refilled
  }

  // Normalize by row sum and store (O layout: lane holds d = lane&15, q = mb+v)
  float inv[8];
#pragma unroll
  for (int v = 0; v < 8; ++v) inv[v] = 1.0f / lane_bcast(lrun, mb + v);
  const int nl = lane & 15;
#pragma unroll
  for (int t = 0; t < 4; ++t) {
#pragma unroll
    for (int v = 0; v < 8; ++v) {
      const int q = qt + mb + v;
      const int d = t * 16 + nl;
      out[(size_t)(b * NQ + q) * so + h * HD + d] = (bf16)(o[t][v] * inv[v]);
    }
  }
}

// ---------------------------------------------------------------------------
// LayerNorm over D=768: one row per block, 256 threads x 3 elements.
// outb != null -> bf16 output; else fp32 to outf.
// ---------------------------------------------------------------------------
__global__ void __launch_bounds__(256)
layernorm_k(const float* __restrict__ x, const float* __restrict__ gamma,
            const float* __restrict__ beta, bf16* __restrict__ outb,
            float* __restrict__ outf) {
  __shared__ float red[256];
  const int tid = threadIdx.x;
  const float* xr = x + (size_t)blockIdx.x * D;
  float v[3];
  float s = 0.f;
#pragma unroll
  for (int j = 0; j < 3; ++j) { v[j] = xr[tid + j * 256]; s += v[j]; }
  red[tid] = s; __syncthreads();
  for (int o = 128; o > 0; o >>= 1) {
    if (tid < o) red[tid] += red[tid + o];
    __syncthreads();
  }
  const float mean = red[0] * (1.0f / (float)D);
  __syncthreads();
  float s2 = 0.f;
#pragma unroll
  for (int j = 0; j < 3; ++j) { const float d0 = v[j] - mean; s2 += d0 * d0; }
  red[tid] = s2; __syncthreads();
  for (int o = 128; o > 0; o >>= 1) {
    if (tid < o) red[tid] += red[tid + o];
    __syncthreads();
  }
  const float rstd = rsqrtf(red[0] * (1.0f / (float)D) + 1e-6f);
  const size_t base = (size_t)blockIdx.x * D;
#pragma unroll
  for (int j = 0; j < 3; ++j) {
    const int c = tid + j * 256;
    const float y = (v[j] - mean) * rstd * gamma[c] + beta[c];
    if (outb) outb[base + c] = (bf16)y;
    else      outf[base + c] = y;
  }
}

// fp32 -> bf16 elementwise convert
__global__ void f2bf(const float* __restrict__ in, bf16* __restrict__ out,
                     int n) {
  const int i = blockIdx.x * 256 + threadIdx.x;
  if (i < n) out[i] = (bf16)in[i];
}

// Build V^T: vt[((b*NHEAD+h)*HD + d)*NC + k] = vsrc[(b*NC+k)*sv + h*HD + d]
// (V section column offset folded into vsrc by caller; head offset added here)
__global__ void build_vt(const bf16* __restrict__ vsrc, int sv,
                         bf16* __restrict__ vt) {
  const int idx = blockIdx.x * 256 + threadIdx.x;  // over BB*NHEAD*HD*NC
  const int k = idx & (NC - 1);
  const int r = idx >> 10;
  const int d = r & (HD - 1);
  const int bh = r >> 6;
  const int h = bh % NHEAD;
  const int b = bh / NHEAD;
  vt[idx] = vsrc[(size_t)(b * NC + k) * sv + h * HD + d];
}

// ---------------------------------------------------------------------------
// Host launcher
// ---------------------------------------------------------------------------
extern "C" void kernel_launch(void* const* d_in, const int* in_sizes, int n_in,
                              void* d_out, int out_size, void* d_ws,
                              size_t ws_size, hipStream_t stream) {
  const float* x      = (const float*)d_in[0];
  const float* ctx    = (const float*)d_in[1];
  const float* ln0_g  = (const float*)d_in[2];
  const float* ln0_b  = (const float*)d_in[3];
  const float* qkv_w  = (const float*)d_in[4];
  const float* qkv_b  = (const float*)d_in[5];
  const float* sap_w  = (const float*)d_in[6];
  const float* sap_b  = (const float*)d_in[7];
  const float* ln1_g  = (const float*)d_in[8];
  const float* ln1_b  = (const float*)d_in[9];
  const float* q_w    = (const float*)d_in[10];
  const float* q_b    = (const float*)d_in[11];
  const float* k_w    = (const float*)d_in[12];
  const float* k_b    = (const float*)d_in[13];
  const float* v_w    = (const float*)d_in[14];
  const float* v_b    = (const float*)d_in[15];
  const float* cap_w  = (const float*)d_in[16];
  const float* cap_b  = (const float*)d_in[17];
  const float* ln2_g  = (const float*)d_in[18];
  const float* ln2_b  = (const float*)d_in[19];
  const float* fc1_w  = (const float*)d_in[20];
  const float* fc1_b  = (const float*)d_in[21];
  const float* fc2_w  = (const float*)d_in[22];
  const float* fc2_b  = (const float*)d_in[23];
  const float* lnf_g  = (const float*)d_in[24];
  const float* lnf_b  = (const float*)d_in[25];

  char* wsp = (char*)d_ws;
  size_t off = 0;
  auto balloc = [&](size_t nbytes) -> void* {
    off = (off + 255) & ~(size_t)255;
    void* p = wsp + off;
    off += nbytes;
    return p;
  };

  const size_t DD = (size_t)D * D;
  bf16* qkv_wb = (bf16*)balloc((size_t)DEPTH * 3 * DD * 2);
  bf16* sap_wb = (bf16*)balloc((size_t)DEPTH * DD * 2);
  bf16* qw_b   = (bf16*)balloc((size_t)DEPTH * DD * 2);
  bf16* kw_b   = (bf16*)balloc((size_t)DEPTH * DD * 2);
  bf16* vw_b   = (bf16*)balloc((size_t)DEPTH * DD * 2);
  bf16* cap_wb = (bf16*)balloc((size_t)DEPTH * DD * 2);
  bf16* fc1_wb = (bf16*)balloc((size_t)DEPTH * (size_t)FF * D * 2);
  bf16* fc2_wb = (bf16*)balloc((size_t)DEPTH * (size_t)FF * D * 2);
  bf16* ctxb   = (bf16*)balloc((size_t)CTX_ROWS * D * 2);
  float* xres  = (float*)balloc((size_t)M_ROWS * D * 4);
  bf16* hbuf   = (bf16*)balloc((size_t)M_ROWS * D * 2);
  bf16* qkvbuf = (bf16*)balloc((size_t)M_ROWS * 3 * D * 2);
  bf16* qbuf   = (bf16*)balloc((size_t)M_ROWS * D * 2);
  bf16* kbuf   = (bf16*)balloc((size_t)CTX_ROWS * D * 2);
  bf16* vbuf   = (bf16*)balloc((size_t)CTX_ROWS * D * 2);
  bf16* vtbuf  = (bf16*)balloc((size_t)BB * NHEAD * HD * NC * 2);
  bf16* attnb  = (bf16*)balloc((size_t)M_ROWS * D * 2);
  bf16* ffbuf  = (bf16*)balloc((size_t)M_ROWS * FF * 2);

  auto conv = [&](const float* src, bf16* dst, size_t n) {
    f2bf<<<dim3((unsigned)((n + 255) / 256)), 256, 0, stream>>>(src, dst,
                                                               (int)n);
  };
  // x -> residual accumulator (fp32)
  hipMemcpyAsync(xres, x, (size_t)M_ROWS * D * 4, hipMemcpyDeviceToDevice,
                 stream);
  conv(ctx, ctxb, (size_t)CTX_ROWS * D);
  conv(qkv_w, qkv_wb, (size_t)DEPTH * 3 * DD);
  conv(sap_w, sap_wb, (size_t)DEPTH * DD);
  conv(q_w, qw_b, (size_t)DEPTH * DD);
  conv(k_w, kw_b, (size_t)DEPTH * DD);
  conv(v_w, vw_b, (size_t)DEPTH * DD);
  conv(cap_w, cap_wb, (size_t)DEPTH * DD);
  conv(fc1_w, fc1_wb, (size_t)DEPTH * (size_t)FF * D);
  conv(fc2_w, fc2_wb, (size_t)DEPTH * (size_t)FF * D);

  auto gemm = [&](const bf16* A, int lda, const bf16* W, const float* bias,
                  bf16* outb, float* outf, int ldo, int M, int N, int K,
                  int epi) {
    dim3 g((unsigned)(N / 64), (unsigned)(M / 128));
    gemm_wmma<<<g, 256, 0, stream>>>(A, lda, W, bias, outb, outf, ldo, K, epi);
  };
  const dim3 agrid(NQ / 128, NHEAD, BB);
  const unsigned vtblocks = (unsigned)((size_t)BB * NHEAD * HD * NC / 256);

  for (int i = 0; i < DEPTH; ++i) {
    // ---- self-attention ----
    layernorm_k<<<M_ROWS, 256, 0, stream>>>(xres, ln0_g + i * D, ln0_b + i * D,
                                            hbuf, nullptr);
    gemm(hbuf, D, qkv_wb + (size_t)i * 3 * DD, qkv_b + (size_t)i * 3 * D,
         qkvbuf, nullptr, 3 * D, M_ROWS, 3 * D, D, 0);
    build_vt<<<vtblocks, 256, 0, stream>>>(qkvbuf + 2 * D, 3 * D, vtbuf);
    attn_wmma<<<agrid, 256, 0, stream>>>(qkvbuf, 3 * D, qkvbuf + D, 3 * D,
                                         vtbuf, attnb, D, 0.125f);
    gemm(attnb, D, sap_wb + (size_t)i * DD, sap_b + (size_t)i * D, nullptr,
         xres, D, M_ROWS, D, D, 1);
    // ---- cross-attention ----
    layernorm_k<<<M_ROWS, 256, 0, stream>>>(xres, ln1_g + i * D, ln1_b + i * D,
                                            hbuf, nullptr);
    gemm(hbuf, D, qw_b + (size_t)i * DD, q_b + (size_t)i * D, qbuf, nullptr, D,
         M_ROWS, D, D, 0);
    gemm(ctxb, D, kw_b + (size_t)i * DD, k_b + (size_t)i * D, kbuf, nullptr, D,
         CTX_ROWS, D, D, 0);
    gemm(ctxb, D, vw_b + (size_t)i * DD, v_b + (size_t)i * D, vbuf, nullptr, D,
         CTX_ROWS, D, D, 0);
    build_vt<<<vtblocks, 256, 0, stream>>>(vbuf, D, vtbuf);
    attn_wmma<<<agrid, 256, 0, stream>>>(qbuf, D, kbuf, D, vtbuf, attnb, D,
                                         0.125f);
    gemm(attnb, D, cap_wb + (size_t)i * DD, cap_b + (size_t)i * D, nullptr,
         xres, D, M_ROWS, D, D, 1);
    // ---- MLP ----
    layernorm_k<<<M_ROWS, 256, 0, stream>>>(xres, ln2_g + i * D, ln2_b + i * D,
                                            hbuf, nullptr);
    gemm(hbuf, D, fc1_wb + (size_t)i * FF * D, fc1_b + (size_t)i * FF, ffbuf,
         nullptr, FF, M_ROWS, FF, D, 2);
    gemm(ffbuf, FF, fc2_wb + (size_t)i * FF * D, fc2_b + (size_t)i * D, nullptr,
         xres, D, M_ROWS, D, FF, 1);
  }
  layernorm_k<<<M_ROWS, 256, 0, stream>>>(xres, lnf_g, lnf_b, nullptr,
                                          (float*)d_out);
}
